// CRF_48266842472984
// MI455X (gfx1250) — matrix operand order repaired and checked by reading, not compiled, not run on hardware
//
#include <hip/hip_runtime.h>
#include <hip/hip_bf16.h>

typedef __attribute__((ext_vector_type(16))) _Float16 v16h;
typedef __attribute__((ext_vector_type(8)))  float    v8f;
typedef __attribute__((ext_vector_type(4)))  float    v4f;

#define TT      52      // tag count (TAGSET + 2)
#define SLEN    256
#define BATCH   256
#define PSTRIDE 68      // padded LDS row stride (floats); 272B rows stay 16B-aligned
#define NEGV    -10000.0f
#define START_T (TT - 2)
#define STOP_T  (TT - 1)
#define LOG2E   1.44269504088896f
#define LN2     0.69314718055995f

// ---------------------------------------------------------------------------
// Kernel 1: scores[s,b,i,j] = feats[b,s,j] + trans[i,j]   (709 MB stream-out)
// Pure store-bandwidth bound (~32us at 23.3 TB/s). NT stores keep the 709MB
// stream out of the 192MB L2 (nothing rereads it). b32 stores forced: output
// base is d_out+1 (Z scalar first) so 16B vector alignment is impossible.
// ---------------------------------------------------------------------------
__global__ void crf_scores_kernel(const float* __restrict__ feats,
                                  const float* __restrict__ trans,
                                  float* __restrict__ out) {
    __shared__ float frow[TT];
    const int sb = blockIdx.x;            // s*BATCH + b
    const int b  = sb & (BATCH - 1);
    const int s  = sb >> 8;
    const float* f = feats + ((size_t)b * SLEN + s) * TT;
    for (int j = threadIdx.x; j < TT; j += blockDim.x) frow[j] = f[j];
    __syncthreads();
    float* o = out + 1 + (size_t)sb * (TT * TT);
    for (int e = threadIdx.x; e < TT * TT; e += blockDim.x) {
        const int j = e % TT;             // e == i*TT + j, matches flat trans
        __builtin_nontemporal_store(trans[e] + frow[j], &o[e]);
    }
}

__global__ void crf_zero_kernel(float* __restrict__ out) {
    if (threadIdx.x == 0) out[0] = 0.0f;
}

// ---------------------------------------------------------------------------
// Kernel 2: forward scan in exp space via WMMA.
//   new_p[b,j] = feats[b,s,j] + m_b + log( sum_i exp(p[b,i]-m_b) * E[i,j] )
// One wave per block, 16 batches per wave. Per step: A(16x64 f16, 2 chunks)
// x E(64x64 f16) -> 8x v_wmma_f32_16x16x32_f16. Branchless body; feats
// double-buffer-prefetched one full step ahead to hide global latency.
// ---------------------------------------------------------------------------
__global__ void __launch_bounds__(32)
crf_forward_kernel(const float* __restrict__ feats,
                   const unsigned char* __restrict__ mask,
                   const float* __restrict__ trans,
                   float* __restrict__ Z) {
    __shared__ float sP[16 * PSTRIDE];      // partition state, 16 batch rows

    const int lane = threadIdx.x & 31;
    const int half = lane >> 4;             // lane group 0/1
    const int row  = lane & 15;
    const int batch_base = blockIdx.x * 16;
    float* P = sP;

    // ---- Preload B fragments: E[k,j] = exp(trans[k,j]), zero-padded.
    // B KxN fragment (contiguous K, same order as A): lane = column N (dup across halves), element h
    // holds K = half*16 + h (+32 for the second K-chunk).
    v16h Bf[2][4];
#pragma unroll
    for (int kc = 0; kc < 2; ++kc) {
#pragma unroll
        for (int nt = 0; nt < 4; ++nt) {
            v16h t;
            const int n = row + nt * 16;
#pragma unroll
            for (int h = 0; h < 16; ++h) {
                const int k = kc * 32 + half * 16 + h;
                float e = 0.0f;
                if (k < TT && n < TT)
                    e = __builtin_amdgcn_exp2f(trans[k * TT + n] * LOG2E);
                t[h] = (_Float16)e;
            }
            Bf[kc][nt] = t;
        }
    }

    // ---- Init partition: p[b,j] = feats[b,0,j] + trans[START,j]
    for (int idx = lane; idx < 16 * 64; idx += 32) {
        const int r = idx >> 6;
        const int j = idx & 63;
        float v = NEGV;
        if (j < TT)
            v = feats[(size_t)(batch_base + r) * SLEN * TT + j]
              + trans[START_T * TT + j];
        P[r * PSTRIDE + j] = v;
    }

    // ---- feats prefetch: pf[nt*8+r] = feats[batch_base + half*8 + r, s, row + nt*16]
    float pf_cur[32], pf_nxt[32];
    auto load_pf = [&](int s, float* dst) {
#pragma unroll
        for (int nt = 0; nt < 4; ++nt) {
            const int j  = row + nt * 16;
            const int jc = (j < TT) ? j : (TT - 1);     // clamped; value unused
#pragma unroll
            for (int r = 0; r < 8; ++r) {
                const int M = half * 8 + r;
                dst[nt * 8 + r] =
                    feats[((size_t)(batch_base + M) * SLEN + s) * TT + jc];
            }
        }
    };
    load_pf(1, pf_cur);

    // ---- Sequential scan (per-wave independent; fully uniform control flow).
    for (int s = 1; s < SLEN; ++s) {
        // Prefetch next step's feats first: consumed one iteration later.
        const int sn = (s + 1 < SLEN) ? (s + 1) : (SLEN - 1);
        load_pf(sn, pf_nxt);

        const bool mk = mask[(size_t)(batch_base + row) * SLEN + s] != 0;
        const unsigned mbits = (unsigned)__ballot(mk);  // bit l = mask of row l&15

        // Row max over tags, lane halves take [0,28) / [28,52), b128 loads.
        const int kst0 = half * 28;
        float mx = -3.4e38f;
#pragma unroll
        for (int k = 0; k < 28; k += 4) {
            const v4f q = *(const v4f*)&P[row * PSTRIDE + kst0 + k];
#pragma unroll
            for (int i = 0; i < 4; ++i) {
                const float v = (kst0 + k + i < TT) ? q[i] : -3.4e38f;
                mx = fmaxf(mx, v);
            }
        }
        mx = fmaxf(mx, __shfl_xor(mx, 16, 32));

        // Build A fragments (contiguous K order, same as B):
        // chunk c, group g: elements g*8+i hold K = c*32 + half*16 + g*8 + i.
        v16h a[2];
#pragma unroll
        for (int c = 0; c < 2; ++c) {
            v16h t;
#pragma unroll
            for (int g = 0; g < 2; ++g) {
                const int kst = c * 32 + half * 16 + g * 8;
                const v4f q0 = *(const v4f*)&P[row * PSTRIDE + kst];
                const v4f q1 = *(const v4f*)&P[row * PSTRIDE + kst + 4];
#pragma unroll
                for (int i = 0; i < 8; ++i) {
                    float x = ((i < 4) ? q0[i] : q1[i - 4]) - mx;
                    x = (kst + i < TT) ? x : -100.0f;   // padded K -> exp ~ 0
                    t[g * 8 + i] = (_Float16)__builtin_amdgcn_exp2f(x * LOG2E);
                }
            }
            a[c] = t;
        }

        // 8 WMMAs (K=64 in 2 chunks, 4 N-tiles), then branchless writeback.
#pragma unroll
        for (int nt = 0; nt < 4; ++nt) {
            v8f acc = {};
            acc = __builtin_amdgcn_wmma_f32_16x16x32_f16(
                    false, a[0], false, Bf[0][nt], (short)0, acc, false, false);
            acc = __builtin_amdgcn_wmma_f32_16x16x32_f16(
                    false, a[1], false, Bf[1][nt], (short)0, acc, false, false);
            const int j = row + nt * 16;                 // j in [0,64): pad cols
#pragma unroll
            for (int r = 0; r < 8; ++r) {
                const int M   = half * 8 + r;            // C/D: M = half*8 + reg
                const float mxM = __shfl(mx, M, 32);     // lane M holds row M max
                const float old = P[M * PSTRIDE + j];
                const float nv  = pf_cur[nt * 8 + r] + mxM
                    + LN2 * __builtin_amdgcn_logf(fmaxf(acc[r], 1e-30f));
                const bool mM = (mbits >> M) & 1;
                P[M * PSTRIDE + j] = mM ? nv : old;      // pad cols: harmless
            }
        }

#pragma unroll
        for (int i = 0; i < 32; ++i) pf_cur[i] = pf_nxt[i];
    }

    // ---- Z contribution: logsumexp over STOP column, summed over 16 rows.
    {
        const int kst0 = half * 28;
        float m2 = -3.4e38f;
#pragma unroll 4
        for (int k = 0; k < 28; ++k) {
            const int kk = kst0 + k;
            const float v = (kk < TT)
                ? P[row * PSTRIDE + kk] + trans[kk * TT + STOP_T] : -3.4e38f;
            m2 = fmaxf(m2, v);
        }
        m2 = fmaxf(m2, __shfl_xor(m2, 16, 32));
        float se = 0.0f;
#pragma unroll 4
        for (int k = 0; k < 28; ++k) {
            const int kk = kst0 + k;
            if (kk < TT)
                se += __builtin_amdgcn_exp2f(
                    (P[row * PSTRIDE + kk] + trans[kk * TT + STOP_T] - m2) * LOG2E);
        }
        se += __shfl_xor(se, 16, 32);
        float part = LN2 * __builtin_amdgcn_logf(fmaxf(se, 1e-30f)) + m2;
        for (int off = 16; off >= 1; off >>= 1)
            part += __shfl_xor(part, off, 32);
        if (lane == 0) atomicAdd(Z, 0.5f * part);        // rows counted twice
    }
}

// ---------------------------------------------------------------------------
extern "C" void kernel_launch(void* const* d_in, const int* in_sizes, int n_in,
                              void* d_out, int out_size, void* d_ws, size_t ws_size,
                              hipStream_t stream) {
    const float*         feats = (const float*)d_in[0];
    const unsigned char* mask  = (const unsigned char*)d_in[1];
    const float*         trans = (const float*)d_in[2];
    float* out = (float*)d_out;

    crf_zero_kernel<<<1, 64, 0, stream>>>(out);
    // scores: one block per (s,b) pair -> 65536 blocks, stream 709 MB out.
    crf_scores_kernel<<<SLEN * BATCH, 256, 0, stream>>>(feats, trans, out);
    // forward scan: 16 independent waves, one per block / WGP.
    crf_forward_kernel<<<16, 32, 0, stream>>>(feats, mask, trans, out);
}
